// TemporalGCN_50749333570013
// MI455X (gfx1250) — matrix operand order repaired and checked by reading, not compiled, not run on hardware
//
#include <hip/hip_runtime.h>

#define Nn 100000
#define Ee 1600000
#define HID 128
#define Gg 256
#define LSTM_H 128
#define OUTD 10
#define EPSV 1e-5f

typedef __attribute__((ext_vector_type(16))) __bf16 v16bf;
typedef __attribute__((ext_vector_type(8)))  __bf16 v8bf;
typedef __attribute__((ext_vector_type(8)))  float  v8f;

static __device__ __forceinline__ unsigned short bfbits(float f) {
    union { float f; unsigned u; } a; a.f = f;
    unsigned r = (a.u + 0x7FFFu + ((a.u >> 16) & 1u)) >> 16;
    return (unsigned short)r;
}
static __device__ __forceinline__ __bf16 f2bf(float f) {
    union { unsigned short s; __bf16 b; } o; o.s = bfbits(f); return o.b;
}

// ---------------- degree / norm ----------------
__global__ void k_deg(const long long* __restrict__ dst, float* __restrict__ deg, int E) {
    int t = blockIdx.x * 256 + threadIdx.x;
    if (t < E) atomicAdd(&deg[(int)dst[t]], 1.0f);
}
__global__ void k_dis(const float* __restrict__ deg, float* __restrict__ dis, int n) {
    int t = blockIdx.x * 256 + threadIdx.x;
    if (t < n) dis[t] = rsqrtf(deg[t] + 1.0f);
}
// ---------------- fp32 -> bf16 convert ----------------
__global__ void k_cvt(const float* __restrict__ x, __bf16* __restrict__ hbf, int n) {
    int t = blockIdx.x * 256 + threadIdx.x;
    if (t < n) hbf[t] = f2bf(x[t]);
}
// ---------------- weight pre-swizzle into WMMA B-fragment order ----------------
// Wp dword index ((nt*4+kt)*32 + lane)*8 + d holds K=(kt*32 + (lane>=16?8:0) + (d>=4?16:0) + 2*(d&3), +1), N=nt*16+(lane&15)
__global__ void k_prepw(const float* __restrict__ W, unsigned* __restrict__ Wp) {
    int t = blockIdx.x * 256 + threadIdx.x;          // 0..8191
    int f = t >> 8;                                   // frag id (nt*4+kt)
    int r = t & 255;
    int lane = r >> 3, d = r & 7;
    int nt = f >> 2, kt = f & 3;
    int n = nt * 16 + (lane & 15);
    int k = kt * 32 + ((lane >> 4) << 3) + ((d >> 2) << 4) + 2 * (d & 3);
    unsigned lo = bfbits(W[k * HID + n]);
    unsigned hi = bfbits(W[(k + 1) * HID + n]);
    Wp[t] = lo | (hi << 16);
}
// ---------------- GEMM: C[N,128] = A[N,128](bf16) * W(bf16 fragments), fp32 acc ----------------
__global__ void __launch_bounds__(256) k_gemm(const __bf16* __restrict__ A,
                                              const unsigned* __restrict__ Wp,
                                              float* __restrict__ C, int nTiles) {
    __shared__ unsigned ldsW[8192];                   // 32 KB, whole 128x128 bf16 W
    const int t = threadIdx.x;
    {
        const uint4* s = reinterpret_cast<const uint4*>(Wp);
        uint4* d = reinterpret_cast<uint4*>(ldsW);
#pragma unroll
        for (int i = 0; i < 8; ++i) d[t + i * 256] = s[t + i * 256];
    }
    __syncthreads();
    const int wave = t >> 5, lane = t & 31;
    const int rowTile = blockIdx.x * 8 + wave;
    if (rowTile >= nTiles) return;

    const int row  = rowTile * 16 + (lane & 15);
    const int kh   = (lane >> 4) << 3;                // 0 or 8 elements (A layout half)
    v8f acc[8];
#pragma unroll
    for (int i = 0; i < 8; ++i) acc[i] = v8f{0.f,0.f,0.f,0.f,0.f,0.f,0.f,0.f};

#pragma unroll
    for (int kt = 0; kt < 4; ++kt) {
        const __bf16* ap = A + (size_t)row * HID + kt * 32 + kh;
        v8bf alo = *reinterpret_cast<const v8bf*>(ap);
        v8bf ahi = *reinterpret_cast<const v8bf*>(ap + 16);
        v16bf a;
#pragma unroll
        for (int i = 0; i < 8; ++i) { a[i] = alo[i]; a[8 + i] = ahi[i]; }
#pragma unroll
        for (int nt = 0; nt < 8; ++nt) {
            const v8bf* bp = reinterpret_cast<const v8bf*>(&ldsW[((nt * 4 + kt) * 32 + lane) * 8]);
            v8bf blo = bp[0], bhi = bp[1];
            v16bf b;
#pragma unroll
            for (int i = 0; i < 8; ++i) { b[i] = blo[i]; b[8 + i] = bhi[i]; }
            acc[nt] = __builtin_amdgcn_wmma_f32_16x16x32_bf16(
                false, a, false, b, (short)0, acc[nt], false, false);
        }
    }
    // C layout: VGPR i -> M = i (lanes 0-15) or 8+i (lanes 16-31); N = nt*16 + (lane&15)
    const int rowBase = rowTile * 16 + ((lane >> 4) << 3);
    const int col = lane & 15;
#pragma unroll
    for (int nt = 0; nt < 8; ++nt)
#pragma unroll
        for (int i = 0; i < 8; ++i)
            C[(size_t)(rowBase + i) * HID + nt * 16 + col] = acc[nt][i];
}
// ---------------- self-loop init: agg = hw*sn + b ----------------
__global__ void k_self(const float* __restrict__ hw, const float* __restrict__ dis,
                       const float* __restrict__ b, float* __restrict__ agg, int n) {
    int t = blockIdx.x * 256 + threadIdx.x;
    if (t < n) {
        int i = t >> 7, c = t & 127;
        float s = dis[i];
        agg[t] = hw[t] * s * s + b[c];
    }
}
// ---------------- edge scatter-add (wave per edge, float4 per lane) ----------------
__global__ void k_edge(const long long* __restrict__ src, const long long* __restrict__ dst,
                       const float* __restrict__ dis, const float* __restrict__ hw,
                       float* __restrict__ agg, int E) {
    int t = blockIdx.x * 256 + threadIdx.x;
    int e = t >> 5, lane = t & 31;
    if (e >= E) return;
    int s = (int)src[e], d = (int)dst[e];
    float en = dis[s] * dis[d];
    const float4 v = *reinterpret_cast<const float4*>(hw + (size_t)s * HID + lane * 4);
    float* ap = agg + (size_t)d * HID + lane * 4;
    atomicAdd(ap + 0, v.x * en);
    atomicAdd(ap + 1, v.y * en);
    atomicAdd(ap + 2, v.z * en);
    atomicAdd(ap + 3, v.w * en);
}
// ---------------- BN stats: per-channel sum/sumsq ----------------
__global__ void k_stats(const float* __restrict__ h, float* __restrict__ sum,
                        float* __restrict__ sq, int n) {
    int c = threadIdx.x;                              // 128 threads
    float s = 0.f, q = 0.f;
    for (int r = blockIdx.x; r < n; r += gridDim.x) {
        float v = h[(size_t)r * HID + c];
        s += v; q += v * v;
    }
    atomicAdd(&sum[c], s);
    atomicAdd(&sq[c], q);
}
__global__ void k_fin(const float* __restrict__ sum, const float* __restrict__ sq,
                      const float* __restrict__ g, const float* __restrict__ be,
                      float* __restrict__ scale, float* __restrict__ shift, float invn) {
    int c = threadIdx.x;
    float mu = sum[c] * invn;
    float var = sq[c] * invn - mu * mu;
    float sc = g[c] * rsqrtf(var + EPSV);
    scale[c] = sc;
    shift[c] = be[c] - mu * sc;
}
// ---------------- BN apply + ReLU, write fp32 and bf16 ----------------
__global__ void k_apply(const float* __restrict__ agg, const float* __restrict__ scale,
                        const float* __restrict__ shift, float* __restrict__ hf,
                        __bf16* __restrict__ hbf, int n) {
    int t = blockIdx.x * 256 + threadIdx.x;
    if (t < n) {
        int c = t & 127;
        float y = fmaxf(agg[t] * scale[c] + shift[c], 0.f);
        hf[t] = y;
        hbf[t] = f2bf(y);
    }
}
// ---------------- graph pooling ----------------
__global__ void k_cnt(const long long* __restrict__ batch, float* __restrict__ cnt, int n) {
    int t = blockIdx.x * 256 + threadIdx.x;
    if (t < n) atomicAdd(&cnt[(int)batch[t]], 1.f);
}
__global__ void k_pool(const float* __restrict__ h, const long long* __restrict__ batch,
                       float* __restrict__ pooled, int n) {
    int t = blockIdx.x * 256 + threadIdx.x;
    if (t < n) {
        int i = t >> 7, c = t & 127;
        atomicAdd(&pooled[(int)batch[i] * HID + c], h[t]);
    }
}
__global__ void k_pooldiv(float* __restrict__ pooled, const float* __restrict__ cnt, int n) {
    int t = blockIdx.x * 256 + threadIdx.x;
    if (t < n) pooled[t] /= fmaxf(cnt[t >> 7], 1.f);
}
// ---------------- LSTM (seq len 1, h0=c0=0) ----------------
__global__ void k_gates(const float* __restrict__ pooled, const float* __restrict__ Wih,
                        const float* __restrict__ bih, const float* __restrict__ bhh,
                        float* __restrict__ gates, int n) {
    int t = blockIdx.x * 256 + threadIdx.x;
    if (t >= n) return;
    int g = t >> 9, j = t & 511;
    const float4* wp = reinterpret_cast<const float4*>(Wih + (size_t)j * HID);
    const float4* pp = reinterpret_cast<const float4*>(pooled + (size_t)g * HID);
    float s = bih[j] + bhh[j];
#pragma unroll 8
    for (int k = 0; k < 32; ++k) {
        float4 w = wp[k], p = pp[k];
        s += w.x * p.x + w.y * p.y + w.z * p.z + w.w * p.w;
    }
    gates[t] = s;
}
static __device__ __forceinline__ float sigm(float x) { return 1.f / (1.f + __expf(-x)); }
__global__ void k_lstm(const float* __restrict__ gates, float* __restrict__ hn, int n) {
    int t = blockIdx.x * 256 + threadIdx.x;
    if (t >= n) return;
    int g = t >> 7, c = t & 127;
    const float* gp = gates + (size_t)g * 512;
    float i = gp[c], f = gp[c + 128], gg = gp[c + 256], o = gp[c + 384];
    (void)f;
    float cc = sigm(i) * tanhf(gg);
    hn[t] = sigm(o) * tanhf(cc);
}
__global__ void k_outp(const float* __restrict__ hn, const float* __restrict__ Wout,
                       const float* __restrict__ bout, float* __restrict__ out, int n) {
    int t = blockIdx.x * 256 + threadIdx.x;
    if (t >= n) return;
    int g = t / OUTD, o = t % OUTD;
    const float4* wp = reinterpret_cast<const float4*>(Wout + (size_t)o * HID);
    const float4* hp = reinterpret_cast<const float4*>(hn + (size_t)g * HID);
    float s = bout[o];
#pragma unroll 8
    for (int k = 0; k < 32; ++k) {
        float4 w = wp[k], h = hp[k];
        s += w.x * h.x + w.y * h.y + w.z * h.z + w.w * h.w;
    }
    out[t] = s;
}

extern "C" void kernel_launch(void* const* d_in, const int* in_sizes, int n_in,
                              void* d_out, int out_size, void* d_ws, size_t ws_size,
                              hipStream_t stream) {
    const float* x    = (const float*)d_in[0];
    const float* W0   = (const float*)d_in[1];
    const float* b0   = (const float*)d_in[2];
    const float* g0   = (const float*)d_in[3];
    const float* be0  = (const float*)d_in[4];
    const float* Wh   = (const float*)d_in[5];
    const float* bh   = (const float*)d_in[6];
    const float* gh   = (const float*)d_in[7];
    const float* beh  = (const float*)d_in[8];
    const float* Wih  = (const float*)d_in[9];
    const float* Whh  = (const float*)d_in[10]; (void)Whh; // h0 = 0 -> only bhh matters
    const float* bih  = (const float*)d_in[11];
    const float* bhh  = (const float*)d_in[12];
    const float* Wout = (const float*)d_in[13];
    const float* bout = (const float*)d_in[14];
    const long long* ei    = (const long long*)d_in[15];
    const long long* batch = (const long long*)d_in[16];
    const long long* srcp = ei;
    const long long* dstp = ei + Ee;

    char* p = (char*)d_ws;
    auto carve = [&](size_t bytes) -> void* {
        void* r = (void*)p;
        p += (bytes + 255) & ~(size_t)255;
        return r;
    };
    float*    deg    = (float*)carve((size_t)Nn * 4);
    float*    dis    = (float*)carve((size_t)Nn * 4);
    __bf16*   hbf    = (__bf16*)carve((size_t)Nn * HID * 2);
    float*    hw     = (float*)carve((size_t)Nn * HID * 4);   // GEMM out, later post-act fp32
    float*    agg    = (float*)carve((size_t)Nn * HID * 4);
    unsigned* Wp     = (unsigned*)carve(32768);
    float*    stats  = (float*)carve(4 * HID * 4);            // sum, sumsq, scale, shift
    float*    pooled = (float*)carve((size_t)Gg * HID * 4);
    float*    cntb   = (float*)carve((size_t)Gg * 4);
    float*    gates  = (float*)carve((size_t)Gg * 512 * 4);
    float*    hn     = (float*)carve((size_t)Gg * HID * 4);
    (void)ws_size; (void)in_sizes; (void)n_in; (void)out_size;

    const int NH = Nn * HID;            // 12,800,000
    const int nTiles = Nn / 16;         // 6250

    hipMemsetAsync(deg, 0, (size_t)Nn * 4, stream);
    k_deg<<<(Ee + 255) / 256, 256, 0, stream>>>(dstp, deg, Ee);
    k_dis<<<(Nn + 255) / 256, 256, 0, stream>>>(deg, dis, Nn);
    k_cvt<<<NH / 256, 256, 0, stream>>>(x, hbf, NH);

    for (int l = 0; l < 3; ++l) {
        const float* W  = (l == 0) ? W0  : Wh  + (size_t)(l - 1) * HID * HID;
        const float* b  = (l == 0) ? b0  : bh  + (size_t)(l - 1) * HID;
        const float* ga = (l == 0) ? g0  : gh  + (size_t)(l - 1) * HID;
        const float* be = (l == 0) ? be0 : beh + (size_t)(l - 1) * HID;

        k_prepw<<<8192 / 256, 256, 0, stream>>>(W, Wp);
        k_gemm<<<(nTiles + 7) / 8, 256, 0, stream>>>(hbf, Wp, hw, nTiles);
        k_self<<<NH / 256, 256, 0, stream>>>(hw, dis, b, agg, NH);
        k_edge<<<(Ee * 32) / 256, 256, 0, stream>>>(srcp, dstp, dis, hw, agg, Ee);
        hipMemsetAsync(stats, 0, 2 * HID * 4, stream);
        k_stats<<<512, 128, 0, stream>>>(agg, stats, stats + HID, Nn);
        k_fin<<<1, 128, 0, stream>>>(stats, stats + HID, ga, be,
                                     stats + 2 * HID, stats + 3 * HID, 1.0f / (float)Nn);
        k_apply<<<NH / 256, 256, 0, stream>>>(agg, stats + 2 * HID, stats + 3 * HID,
                                              hw, hbf, NH);
    }

    hipMemsetAsync(pooled, 0, (size_t)Gg * HID * 4, stream);
    hipMemsetAsync(cntb, 0, (size_t)Gg * 4, stream);
    k_cnt<<<(Nn + 255) / 256, 256, 0, stream>>>(batch, cntb, Nn);
    k_pool<<<NH / 256, 256, 0, stream>>>(hw, batch, pooled, NH);
    k_pooldiv<<<(Gg * HID) / 256, 256, 0, stream>>>(pooled, cntb, Gg * HID);
    k_gates<<<(Gg * 512) / 256, 256, 0, stream>>>(pooled, Wih, bih, bhh, gates, Gg * 512);
    k_lstm<<<(Gg * HID) / 256, 256, 0, stream>>>(gates, hn, Gg * HID);
    k_outp<<<(Gg * OUTD + 255) / 256, 256, 0, stream>>>(hn, Wout, bout, (float*)d_out, Gg * OUTD);
}
